// PinnLayer_38044820308033
// MI455X (gfx1250) — matrix operand order repaired and checked by reading, not compiled, not run on hardware
//
#include <hip/hip_runtime.h>

// ---------------------------------------------------------------------------
// PINN: 3x tanh MLP (H=512) + nested t-JVPs + sin-connectivity physics.
// B=1024, N=H=512. GEMMs on v_wmma_f32_16x16x32_f16 (f32 accum).
// Per-wave tile 16x32 (6 accumulators for the 3 JVP streams), weight panel
// staged in LDS (pitch-padded, bank-conflict-free ds_load_b128).
// ---------------------------------------------------------------------------

#define T_MAX_N 10.0f

namespace {
constexpr int BB = 1024;
constexpr int NN = 512;
constexpr int HH = 512;
constexpr int LPITCH = 520;  // 32-col LDS panel row pitch (halves); 1040B % 256B == 16B
}

typedef __attribute__((ext_vector_type(16))) _Float16 v16h;
typedef __attribute__((ext_vector_type(8)))  _Float16 v8h;
typedef __attribute__((ext_vector_type(8)))  float    v8f;

#define WMMA_F16(a, b, c) \
  __builtin_amdgcn_wmma_f32_16x16x32_f16(false, (a), false, (b), (short)0, (c), false, false)

// A fragment: row-major [M x lda] f16 tile at (row0, k0).
// 16-bit A 16x32 layout: lanes 0-15 -> M=lane, K in {0..7} U {16..23};
// lanes 16-31 -> M=lane-16, K in {8..15} U {24..31}. Two b128 loads per lane.
__device__ __forceinline__ v16h frag_a(const _Float16* __restrict__ A, int lda,
                                       int row0, int k0, int lane) {
  const int m = lane & 15, g = lane >> 4;
  const _Float16* p = A + (size_t)(row0 + m) * lda + k0 + g * 8;
  v8h lo = *(const v8h*)p;         // K = g*8 + 0..7
  v8h hi = *(const v8h*)(p + 16);  // K = 16 + g*8 + 0..7
  v16h r;
#pragma unroll
  for (int h = 0; h < 8; ++h) { r[h] = lo[h]; r[8 + h] = hi[h]; }
  return r;
}

// B fragment from the LDS-staged transposed panel (pitch LPITCH).
// 16-bit B 32x16 layout: lanes 0-15 -> N=lane, K in {0..7} U {16..23}; lanes 16-31 -> K in {8..15} U {24..31}.
__device__ __forceinline__ v16h frag_b_lds(const _Float16* lw, int n0, int k0, int lane) {
  const int n = lane & 15, g = lane >> 4;
  const _Float16* p = lw + (n0 + n) * LPITCH + k0 + g * 8;
  v8h lo = *(const v8h*)p;
  v8h hi = *(const v8h*)(p + 16);
  v16h r;
#pragma unroll
  for (int h = 0; h < 8; ++h) { r[h] = lo[h]; r[8 + h] = hi[h]; }
  return r;
}

// Cooperative copy of a 32-column weight panel (Bt rows c0S..c0S+31, K=512)
// into LDS with padded pitch. 2048 b128 chunks / 256 threads = 8 each.
__device__ __forceinline__ void load_panel(const _Float16* __restrict__ Bt, int c0S,
                                           _Float16* lw) {
#pragma unroll
  for (int i = 0; i < 8; ++i) {
    int c = threadIdx.x + i * 256;   // 0..2047
    int n = c >> 6, kc = c & 63;     // row in panel, 8-half chunk in row
    *(v8h*)(lw + n * LPITCH + kc * 8) =
        *(const v8h*)(Bt + (size_t)(c0S + n) * 512 + kc * 8);
  }
  __syncthreads();
}

// Tile coords: 16 column super-tiles (32 wide) x 64 row tiles -> 1024 waves,
// 128 blocks of 8 waves. All waves of a block share one LDS weight panel.
__device__ __forceinline__ void tile_coords2(int& r0, int& c0S, int& lane) {
  lane = threadIdx.x & 31;
  const int cs = blockIdx.x & 15;
  const int rg = blockIdx.x >> 4;
  r0  = (rg * 8 + (threadIdx.x >> 5)) * 16;
  c0S = cs * 32;
}

// ------------------------------- prep kernels ------------------------------

__global__ void k_prep_inputs(const float* __restrict__ t, const float* __restrict__ p,
                              const float* __restrict__ plo, const float* __restrict__ pup,
                              _Float16* __restrict__ pn16, float* __restrict__ tn) {
  int idx = blockIdx.x * blockDim.x + threadIdx.x;  // B*N threads exactly
  int b = idx >> 9, j = idx & 511;
  float lo = plo[j], up = pup[j];
  float v = (up == lo) ? 0.f : (2.f * (p[idx] - lo) / (up - lo) - 1.f);
  pn16[idx] = (_Float16)v;
  if (j == 0) tn[b] = 2.f * t[b] / T_MAX_N - 1.f;
}

// Split W0 [513 x 512]: row 0 -> w0t (f32), rows 1.. -> transposed f16 [512x512]
__global__ void k_prep_w0(const float* __restrict__ W0, _Float16* __restrict__ W0t,
                          float* __restrict__ w0t) {
  int idx = blockIdx.x * blockDim.x + threadIdx.x;  // 512*512
  int n = idx >> 9, k = idx & 511;
  W0t[(size_t)n * 512 + k] = (_Float16)W0[(size_t)(k + 1) * 512 + n];
  if (k == 0) w0t[n] = W0[n];
}

__global__ void k_tcast(const float* __restrict__ W, _Float16* __restrict__ Wt) {
  int idx = blockIdx.x * blockDim.x + threadIdx.x;  // 512*512
  int n = idx >> 9, k = idx & 511;
  Wt[(size_t)n * 512 + k] = (_Float16)W[(size_t)k * 512 + n];
}

// Straight cast (lambda_b row-major IS the transposed conn B-matrix).
__global__ void k_cast16(const float* __restrict__ X, _Float16* __restrict__ Y) {
  int i = blockIdx.x * blockDim.x + threadIdx.x;
  Y[i] = (_Float16)X[i];
}

// ------------------------------- GEMM kernels ------------------------------

// Layer 0: single GEMM pn@W0p; epilogue adds rank-1 t-term and builds JVP seeds
// (z0' = (2/T)*W0[0,:] is batch-independent; z0'' = 0).
__global__ void __launch_bounds__(256) k_layer0(
    const _Float16* __restrict__ pn16, const float* __restrict__ tn,
    const _Float16* __restrict__ W0t, const float* __restrict__ w0t,
    const float* __restrict__ b0,
    _Float16* __restrict__ Ov, _Float16* __restrict__ O1, _Float16* __restrict__ O2) {
  __shared__ _Float16 lw[32 * LPITCH];
  int r0, c0S, lane; tile_coords2(r0, c0S, lane);
  load_panel(W0t, c0S, lw);
  v8f acc[2] = {v8f{}, v8f{}};
  for (int k0 = 0; k0 < 512; k0 += 32) {
    v16h a = frag_a(pn16, 512, r0, k0, lane);
    v16h b0f = frag_b_lds(lw, 0, k0, lane);
    v16h b1f = frag_b_lds(lw, 16, k0, lane);
    acc[0] = WMMA_F16(a, b0f, acc[0]);
    acc[1] = WMMA_F16(a, b1f, acc[1]);
  }
  const int g = lane >> 4;
#pragma unroll
  for (int j = 0; j < 2; ++j) {
    const int col = c0S + j * 16 + (lane & 15);
    const float wt = w0t[col];
    const float zp = wt * (2.0f / T_MAX_N);
    const float bc = b0[col];
#pragma unroll
    for (int r = 0; r < 8; ++r) {
      int row = r0 + r + 8 * g;
      float z = acc[j][r] + tn[row] * wt + bc;
      float a = tanhf(z);
      float s = 1.f - a * a;
      float ap = s * zp;                 // a' = (1-a^2) z'
      float app = -2.f * a * ap * zp;    // a'' (z''=0)
      size_t o = (size_t)row * 512 + col;
      Ov[o] = (_Float16)a; O1[o] = (_Float16)ap; O2[o] = (_Float16)app;
    }
  }
}

// Hidden layer: 6 fused WMMAs per K-step (value/1st/2nd tangent x 2 col tiles)
// sharing both the A fragments and the LDS weight panel.
__global__ void __launch_bounds__(256) k_layer(
    const _Float16* __restrict__ Av, const _Float16* __restrict__ A1,
    const _Float16* __restrict__ A2,
    const _Float16* __restrict__ Wt, const float* __restrict__ bias,
    _Float16* __restrict__ Ov, _Float16* __restrict__ O1, _Float16* __restrict__ O2) {
  __shared__ _Float16 lw[32 * LPITCH];
  int r0, c0S, lane; tile_coords2(r0, c0S, lane);
  load_panel(Wt, c0S, lw);
  v8f acc[2][3] = {{v8f{}, v8f{}, v8f{}}, {v8f{}, v8f{}, v8f{}}};
  for (int k0 = 0; k0 < 512; k0 += 32) {
    v16h av  = frag_a(Av, 512, r0, k0, lane);
    v16h a1f = frag_a(A1, 512, r0, k0, lane);
    v16h a2f = frag_a(A2, 512, r0, k0, lane);
#pragma unroll
    for (int j = 0; j < 2; ++j) {
      v16h b = frag_b_lds(lw, j * 16, k0, lane);
      acc[j][0] = WMMA_F16(av,  b, acc[j][0]);
      acc[j][1] = WMMA_F16(a1f, b, acc[j][1]);
      acc[j][2] = WMMA_F16(a2f, b, acc[j][2]);
    }
  }
  const int g = lane >> 4;
#pragma unroll
  for (int j = 0; j < 2; ++j) {
    const int col = c0S + j * 16 + (lane & 15);
    const float bc = bias[col];
#pragma unroll
    for (int r = 0; r < 8; ++r) {
      int row = r0 + r + 8 * g;
      float z = acc[j][0][r] + bc, zp = acc[j][1][r], zpp = acc[j][2][r];
      float a = tanhf(z);
      float s = 1.f - a * a;
      float ap = s * zp;                         // a' = (1-a^2) z'
      float app = s * zpp - 2.f * a * ap * zp;   // a'' = (1-a^2)z'' - 2 a a' z'
      size_t o = (size_t)row * 512 + col;
      Ov[o] = (_Float16)a; O1[o] = (_Float16)ap; O2[o] = (_Float16)app;
    }
  }
}

// Output layer (linear): emits out, out_t (d_out), out_tt (ws), sin/cos f16.
__global__ void __launch_bounds__(256) k_out(
    const _Float16* __restrict__ Av, const _Float16* __restrict__ A1,
    const _Float16* __restrict__ A2,
    const _Float16* __restrict__ Wt, const float* __restrict__ bias,
    float* __restrict__ out, float* __restrict__ outt, float* __restrict__ outtt,
    _Float16* __restrict__ S16, _Float16* __restrict__ C16) {
  __shared__ _Float16 lw[32 * LPITCH];
  int r0, c0S, lane; tile_coords2(r0, c0S, lane);
  load_panel(Wt, c0S, lw);
  v8f acc[2][3] = {{v8f{}, v8f{}, v8f{}}, {v8f{}, v8f{}, v8f{}}};
  for (int k0 = 0; k0 < 512; k0 += 32) {
    v16h av  = frag_a(Av, 512, r0, k0, lane);
    v16h a1f = frag_a(A1, 512, r0, k0, lane);
    v16h a2f = frag_a(A2, 512, r0, k0, lane);
#pragma unroll
    for (int j = 0; j < 2; ++j) {
      v16h b = frag_b_lds(lw, j * 16, k0, lane);
      acc[j][0] = WMMA_F16(av,  b, acc[j][0]);
      acc[j][1] = WMMA_F16(a1f, b, acc[j][1]);
      acc[j][2] = WMMA_F16(a2f, b, acc[j][2]);
    }
  }
  const int g = lane >> 4;
#pragma unroll
  for (int j = 0; j < 2; ++j) {
    const int col = c0S + j * 16 + (lane & 15);
    const float bc = bias[col];
#pragma unroll
    for (int r = 0; r < 8; ++r) {
      int row = r0 + r + 8 * g;
      size_t o = (size_t)row * 512 + col;
      float v = acc[j][0][r] + bc;
      out[o] = v; outt[o] = acc[j][1][r]; outtt[o] = acc[j][2][r];
      S16[o] = (_Float16)sinf(v);
      C16[o] = (_Float16)cosf(v);
    }
  }
}

// Connectivity + physics: conn = sin(d_i)*(cos(d)@Lb^T) - cos(d_i)*(sin(d)@Lb^T)
// physics = lam_m*out_tt + lam_d*out_t + conn - power
__global__ void __launch_bounds__(256) k_conn(
    const _Float16* __restrict__ S16, const _Float16* __restrict__ C16,
    const _Float16* __restrict__ lb16,
    const float* __restrict__ out, const float* __restrict__ outt,
    const float* __restrict__ outtt,
    const float* __restrict__ lam_m, const float* __restrict__ lam_d,
    const float* __restrict__ power, float* __restrict__ phys) {
  __shared__ _Float16 lw[32 * LPITCH];
  int r0, c0S, lane; tile_coords2(r0, c0S, lane);
  load_panel(lb16, c0S, lw);  // B[j][i]=lambda_b[i][j] -> Bt storage == lambda_b
  v8f aC[2] = {v8f{}, v8f{}}, aS[2] = {v8f{}, v8f{}};
  for (int k0 = 0; k0 < 512; k0 += 32) {
    v16h cf = frag_a(C16, 512, r0, k0, lane);
    v16h sf = frag_a(S16, 512, r0, k0, lane);
#pragma unroll
    for (int j = 0; j < 2; ++j) {
      v16h b = frag_b_lds(lw, j * 16, k0, lane);
      aC[j] = WMMA_F16(cf, b, aC[j]);
      aS[j] = WMMA_F16(sf, b, aS[j]);
    }
  }
  const int g = lane >> 4;
#pragma unroll
  for (int j = 0; j < 2; ++j) {
    const int col = c0S + j * 16 + (lane & 15);
    const float lm = lam_m[col], ld = lam_d[col];
#pragma unroll
    for (int r = 0; r < 8; ++r) {
      int row = r0 + r + 8 * g;
      size_t o = (size_t)row * 512 + col;
      float d = out[o];
      float sv = sinf(d), cvv = cosf(d);  // f32 recompute for elementwise part
      phys[o] = lm * outtt[o] + ld * outt[o] + sv * aC[j][r] - cvv * aS[j][r] - power[o];
    }
  }
}

// ------------------------------- launcher ----------------------------------

extern "C" void kernel_launch(void* const* d_in, const int* in_sizes, int n_in,
                              void* d_out, int out_size, void* d_ws, size_t ws_size,
                              hipStream_t stream) {
  (void)in_sizes; (void)n_in; (void)out_size; (void)ws_size;
  const float* t_in  = (const float*)d_in[0];
  const float* p_in  = (const float*)d_in[1];
  const float* W0    = (const float*)d_in[2];
  const float* b0    = (const float*)d_in[3];
  const float* W1    = (const float*)d_in[4];
  const float* b1    = (const float*)d_in[5];
  const float* W2    = (const float*)d_in[6];
  const float* b2    = (const float*)d_in[7];
  const float* Wout  = (const float*)d_in[8];
  const float* bout  = (const float*)d_in[9];
  const float* lam_m = (const float*)d_in[10];
  const float* lam_d = (const float*)d_in[11];
  const float* lam_b = (const float*)d_in[12];
  const float* p_lo  = (const float*)d_in[13];
  const float* p_up  = (const float*)d_in[14];

  float* out  = (float*)d_out;
  float* outt = out + (size_t)BB * NN;
  float* phys = out + 2 * (size_t)BB * NN;

  // Carve workspace (256B-aligned), total ~14.2 MB.
  char* w = (char*)d_ws;
  auto alloc = [&](size_t bytes) -> void* {
    void* p = (void*)w; w += (bytes + 255) & ~(size_t)255; return p;
  };
  _Float16* pn16  = (_Float16*)alloc((size_t)BB * NN * 2);
  float*    tn    = (float*)   alloc((size_t)BB * 4);
  float*    w0t   = (float*)   alloc((size_t)HH * 4);
  _Float16* W0t   = (_Float16*)alloc((size_t)HH * HH * 2);
  _Float16* W1t   = (_Float16*)alloc((size_t)HH * HH * 2);
  _Float16* W2t   = (_Float16*)alloc((size_t)HH * HH * 2);
  _Float16* Wot   = (_Float16*)alloc((size_t)HH * NN * 2);
  _Float16* lb16  = (_Float16*)alloc((size_t)NN * NN * 2);
  _Float16* Xa    = (_Float16*)alloc((size_t)BB * HH * 2);  // act set A
  _Float16* Xa1   = (_Float16*)alloc((size_t)BB * HH * 2);
  _Float16* Xa2   = (_Float16*)alloc((size_t)BB * HH * 2);
  _Float16* Xb    = (_Float16*)alloc((size_t)BB * HH * 2);  // act set B
  _Float16* Xb1   = (_Float16*)alloc((size_t)BB * HH * 2);
  _Float16* Xb2   = (_Float16*)alloc((size_t)BB * HH * 2);
  float*    outtt = (float*)   alloc((size_t)BB * NN * 4);
  _Float16* S16   = (_Float16*)alloc((size_t)BB * NN * 2);
  _Float16* C16   = (_Float16*)alloc((size_t)BB * NN * 2);

  // ---- prep ----
  k_prep_inputs<<<(BB * NN) / 256, 256, 0, stream>>>(t_in, p_in, p_lo, p_up, pn16, tn);
  k_prep_w0<<<(HH * HH) / 256, 256, 0, stream>>>(W0, W0t, w0t);
  k_tcast<<<(HH * HH) / 256, 256, 0, stream>>>(W1, W1t);
  k_tcast<<<(HH * HH) / 256, 256, 0, stream>>>(W2, W2t);
  k_tcast<<<(HH * NN) / 256, 256, 0, stream>>>(Wout, Wot);
  k_cast16<<<(NN * NN) / 256, 256, 0, stream>>>(lam_b, lb16);

  // ---- MLP + JVPs ---- (1024 tile-waves -> 128 blocks of 8 waves)
  const int GB = 128, TB = 256;
  k_layer0<<<GB, TB, 0, stream>>>(pn16, tn, W0t, w0t, b0, Xa, Xa1, Xa2);
  k_layer <<<GB, TB, 0, stream>>>(Xa, Xa1, Xa2, W1t, b1, Xb, Xb1, Xb2);
  k_layer <<<GB, TB, 0, stream>>>(Xb, Xb1, Xb2, W2t, b2, Xa, Xa1, Xa2);
  k_out   <<<GB, TB, 0, stream>>>(Xa, Xa1, Xa2, Wot, bout, out, outt, outtt, S16, C16);

  // ---- connectivity + physics residual ----
  k_conn  <<<GB, TB, 0, stream>>>(S16, C16, lb16, out, outt, outtt,
                                  lam_m, lam_d, p_in, phys);
}
